// gcnWeight_84018150245104
// MI455X (gfx1250) — matrix operand order repaired and checked by reading, not compiled, not run on hardware
//
#include <hip/hip_runtime.h>
#include <hip/hip_bf16.h>

// ---------------------------------------------------------------------------
// Problem constants (from reference):
//   x: [B=64, C=32, N=1024, L=12] fp32 ; A, weight: [1024,1024] fp32
//   mlp_w: [32, 96] ; mlp_b: [32]
//   M = B*C*L = 24576 rows for the diffusion GEMMs, K = N = 1024
// ---------------------------------------------------------------------------

typedef __attribute__((ext_vector_type(16))) __bf16 bf16x16;
typedef __attribute__((ext_vector_type(8)))  __bf16 bf16x8;
typedef __attribute__((ext_vector_type(8)))  float  f32x8;

#define GB  64
#define GC  32
#define GN  1024
#define GL  12
#define GCO 32
#define GM  (GB * GC * GL)   // 24576

// ---------------------------------------------------------------------------
// Kernel 1: waT[w][v] = bf16( (weight @ A)[v][w] )   -- fp32 tiled GEMM
// grid (64,64), block (16,16).  ~2 GFLOP, negligible; fp32 keeps wa accurate
// before the single rounding to bf16.
// ---------------------------------------------------------------------------
__global__ void waT_kernel(const float* __restrict__ W,
                           const float* __restrict__ A,
                           __bf16* __restrict__ waT) {
  __shared__ float sW[16][17];
  __shared__ float sA[16][17];
  const int tx = threadIdx.x, ty = threadIdx.y;
  const int i = blockIdx.y * 16 + ty;   // row of wa (v)
  const int j = blockIdx.x * 16 + tx;   // col of wa (w)
  float acc = 0.f;
  for (int kk = 0; kk < GN; kk += 16) {
    sW[ty][tx] = W[(size_t)i * GN + kk + tx];
    sA[ty][tx] = A[(size_t)(kk + ty) * GN + j];
    __syncthreads();
#pragma unroll
    for (int k = 0; k < 16; ++k) acc += sW[ty][k] * sA[k][tx];
    __syncthreads();
  }
  waT[(size_t)j * GN + i] = (__bf16)acc;   // transposed store: waT[w][v]
}

// ---------------------------------------------------------------------------
// Kernel 2: xT[m][v] = bf16(x[b][c][v][l]),  m = (b*C + c)*L + l
// grid (B*C, N/256), block 256.  LDS staging keeps both sides coalesced.
// ---------------------------------------------------------------------------
__global__ void xT_kernel(const float* __restrict__ x,
                          __bf16* __restrict__ xT) {
  __shared__ float tile[256 * 13];   // 256 v  x 12 l, padded to 13
  const int bc = blockIdx.x;         // b*C + c
  const int v0 = blockIdx.y * 256;
  const float* src = x + ((size_t)bc * GN + v0) * GL;
  for (int e = threadIdx.x; e < 256 * GL; e += 256) {
    int v = e / GL, l = e % GL;
    tile[v * 13 + l] = src[e];                 // contiguous global read
  }
  __syncthreads();
  for (int e = threadIdx.x; e < 256 * GL; e += 256) {
    int l = e >> 8;                            // e / 256
    int v = e & 255;
    size_t m = (size_t)bc * GL + l;
    xT[m * GN + v0 + v] = (__bf16)tile[v * 13 + l];  // contiguous global write
  }
}

// ---------------------------------------------------------------------------
// Kernel 3: bf16 WMMA GEMM:  D[m][w] = sum_v Amat[m][v] * wa[v][w]
//   Amat : bf16 [M][1024] row-major
//   Bt   : bf16 [1024 w][1024 v]  (wa transposed -> K contiguous)
//   Dbf  : bf16 [M][1024] (input to the next GEMM), written iff WRITE_BF
//   Dnat : fp32 scattered into native x layout [b][c][v][l] (for the mlp)
// Block tile 256(M) x 128(N) x 32(K); 256 threads = 8 waves arranged 4(M)x2(N),
// each wave owns a 64x64 output tile -> 16 v_wmma_f32_16x16x32_bf16 per K-step
// against 16 ds_load_b128 + 6 ds_store_b128 (2x the matrix-pipe density of a
// 32x64 wave tile).  LDS: 256x40 + 128x40 bf16 = 30 KB.
// ---------------------------------------------------------------------------
template <bool WRITE_BF>
__global__ void __launch_bounds__(256)
gemm_bf16_kernel(const __bf16* __restrict__ Amat,
                 const __bf16* __restrict__ Bt,
                 __bf16* __restrict__ Dbf,
                 float* __restrict__ Dnat) {
  __shared__ __align__(16) __bf16 sA[256 * 40];   // 256 rows x 32 K (+8 pad)
  __shared__ __align__(16) __bf16 sB[128 * 40];   // 128 rows x 32 K (+8 pad)

  const int tid    = threadIdx.x;
  const int lane   = tid & 31;
  const int waveId = tid >> 5;
  const int half   = lane >> 4;      // 0: lanes 0-15, 1: lanes 16-31
  const int l16    = lane & 15;

  const int waveM = waveId & 3;      // 0..3 -> M offset waveM*64
  const int waveN = waveId >> 2;     // 0..1 -> N offset waveN*64
  const int blockN = blockIdx.x * 128;
  const int blockM = blockIdx.y * 256;

  // ---- global->LDS staging: chunks of 8 bf16 (16 B) -----------------------
  // A tile: 256 rows x 4 chunks = 1024 chunks -> 4 per thread
  // B tile: 128 rows x 4 chunks =  512 chunks -> 2 per thread
  const __bf16* gA[4];
  __bf16*       lA[4];
#pragma unroll
  for (int i = 0; i < 4; ++i) {
    const int chunk = tid + 256 * i;
    const int r = chunk >> 2, k = (chunk & 3) * 8;
    gA[i] = Amat + (size_t)(blockM + r) * GN + k;
    lA[i] = &sA[r * 40 + k];
  }
  const __bf16* gB[2];
  __bf16*       lB[2];
#pragma unroll
  for (int i = 0; i < 2; ++i) {
    const int chunk = tid + 256 * i;
    const int r = chunk >> 2, k = (chunk & 3) * 8;
    gB[i] = Bt + (size_t)(blockN + r) * GN + k;
    lB[i] = &sB[r * 40 + k];
  }

  f32x8 acc[4][4];
#pragma unroll
  for (int i = 0; i < 4; ++i)
#pragma unroll
    for (int j = 0; j < 4; ++j)
#pragma unroll
      for (int r = 0; r < 8; ++r) acc[i][j][r] = 0.f;

  for (int kk = 0; kk < GN; kk += 32) {
    uint4 stA[4], stB[2];
#pragma unroll
    for (int i = 0; i < 4; ++i) stA[i] = *(const uint4*)(gA[i] + kk);
#pragma unroll
    for (int i = 0; i < 2; ++i) stB[i] = *(const uint4*)(gB[i] + kk);
    __syncthreads();                 // previous-iteration LDS readers done
#pragma unroll
    for (int i = 0; i < 4; ++i) *(uint4*)lA[i] = stA[i];
#pragma unroll
    for (int i = 0; i < 2; ++i) *(uint4*)lB[i] = stB[i];
    __syncthreads();                 // tiles ready

    // A fragments: lane<16 holds K {K0..K0+7, K0+16..K0+23} with K0 = half*8
    bf16x16 afr[4];
#pragma unroll
    for (int ms = 0; ms < 4; ++ms) {
      const __bf16* p = &sA[(waveM * 64 + ms * 16 + l16) * 40 + half * 8];
      bf16x8 lo = *(const bf16x8*)p;
      bf16x8 hi = *(const bf16x8*)(p + 16);
      afr[ms] = __builtin_shufflevector(lo, hi, 0, 1, 2, 3, 4, 5, 6, 7,
                                        8, 9, 10, 11, 12, 13, 14, 15);
    }
    // B fragments: column n = l16, K {K0..K0+7, K0+16..K0+23} with K0 = half*8
    bf16x16 bfr[4];
#pragma unroll
    for (int ns = 0; ns < 4; ++ns) {
      const __bf16* p = &sB[(waveN * 64 + ns * 16 + l16) * 40 + half * 8];
      bf16x8 lo = *(const bf16x8*)p;
      bf16x8 hi = *(const bf16x8*)(p + 16);
      bfr[ns] = __builtin_shufflevector(lo, hi, 0, 1, 2, 3, 4, 5, 6, 7,
                                        8, 9, 10, 11, 12, 13, 14, 15);
    }
#pragma unroll
    for (int ms = 0; ms < 4; ++ms)
#pragma unroll
      for (int ns = 0; ns < 4; ++ns)
        acc[ms][ns] = __builtin_amdgcn_wmma_f32_16x16x32_bf16(
            false, afr[ms], false, bfr[ns], (short)0, acc[ms][ns],
            false, false);
  }

  // ---- epilogue: D layout = VGPR r -> M = r + half*8, N = l16 -------------
#pragma unroll
  for (int ms = 0; ms < 4; ++ms) {
    const int tileM = blockM + waveM * 64 + ms * 16;
#pragma unroll
    for (int ns = 0; ns < 4; ++ns) {
      const int col = blockN + waveN * 64 + ns * 16 + l16;
#pragma unroll
      for (int r = 0; r < 8; ++r) {
        const int m = tileM + r + half * 8;
        const float val = acc[ms][ns][r];
        if (WRITE_BF) Dbf[(size_t)m * GN + col] = (__bf16)val;
        // m = (b*C + c)*L + l  ->  native index ((b*C+c)*N + col)*L + l
        const int bcl = m / GL, l = m - bcl * GL;
        Dnat[((size_t)bcl * GN + col) * GL + l] = val;
      }
    }
  }
}

// ---------------------------------------------------------------------------
// Kernel 4: out[b][o][v][l] = mlp_b[o] + sum_c ( w[o][c]    * x [b][c][v][l]
//                                              + w[o][32+c] * x1[b][c][v][l]
//                                              + w[o][64+c] * x2[b][c][v][l] )
// One thread per (b,v,l) position; all reads/writes coalesced, weight reads
// uniform (scalarizable). 3072 blocks x 256 threads; memory-bound pass.
// ---------------------------------------------------------------------------
__global__ void mlp_kernel(const float* __restrict__ x,
                           const float* __restrict__ x1,
                           const float* __restrict__ x2,
                           const float* __restrict__ mw,
                           const float* __restrict__ mb,
                           float* __restrict__ out) {
  const size_t p = (size_t)blockIdx.x * 256 + threadIdx.x;  // over B*N*L
  const int l  = (int)(p % GL);
  const size_t bv = p / GL;
  const int v  = (int)(bv % GN);
  const int b  = (int)(bv / GN);
  const size_t cs   = (size_t)GN * GL;                      // channel stride
  const size_t base = ((size_t)b * GC * GN + v) * GL + l;   // c = 0

  float acc[GCO];
#pragma unroll
  for (int o = 0; o < GCO; ++o) acc[o] = mb[o];

#pragma unroll 4
  for (int c = 0; c < GC; ++c) {
    const float h0 = x [base + (size_t)c * cs];
    const float h1 = x1[base + (size_t)c * cs];
    const float h2 = x2[base + (size_t)c * cs];
#pragma unroll
    for (int o = 0; o < GCO; ++o) {
      acc[o] = fmaf(mw[o * 96 + c],      h0,
               fmaf(mw[o * 96 + 32 + c], h1,
               fmaf(mw[o * 96 + 64 + c], h2, acc[o])));
    }
  }

  const size_t ob = ((size_t)b * GCO * GN + v) * GL + l;
#pragma unroll
  for (int o = 0; o < GCO; ++o) out[ob + (size_t)o * cs] = acc[o];
}

// ---------------------------------------------------------------------------
// Host launcher
// ---------------------------------------------------------------------------
extern "C" void kernel_launch(void* const* d_in, const int* in_sizes, int n_in,
                              void* d_out, int out_size, void* d_ws, size_t ws_size,
                              hipStream_t stream) {
  const float* x  = (const float*)d_in[0];   // [64,32,1024,12]
  const float* A  = (const float*)d_in[1];   // [1024,1024]
  const float* W  = (const float*)d_in[2];   // [1024,1024]
  const float* mw = (const float*)d_in[3];   // [32,96]
  const float* mb = (const float*)d_in[4];   // [32]
  float* out = (float*)d_out;

  char* ws = (char*)d_ws;
  size_t off = 0;
  __bf16* waT = (__bf16*)(ws + off); off += (size_t)GN * GN * sizeof(__bf16);       //   2 MB
  __bf16* xT  = (__bf16*)(ws + off); off += (size_t)GM * GN * sizeof(__bf16);       //  50 MB
  __bf16* x1t = (__bf16*)(ws + off); off += (size_t)GM * GN * sizeof(__bf16);       //  50 MB
  float*  x1n = (float*) (ws + off); off += (size_t)GM * GN * sizeof(float);        // 100 MB
  float*  x2n = (float*) (ws + off); off += (size_t)GM * GN * sizeof(float);        // 100 MB
  (void)ws_size; (void)in_sizes; (void)n_in; (void)out_size;

  // 1) waT = bf16( (weight @ A)^T )
  waT_kernel<<<dim3(GN / 16, GN / 16), dim3(16, 16), 0, stream>>>(W, A, waT);

  // 2) xT = bf16 transpose of x into [M, N]
  xT_kernel<<<dim3(GB * GC, GN / 256), 256, 0, stream>>>(x, xT);

  // 3) x1 = xT @ wa   (bf16 WMMA, fp32 accumulate) -> x1t (bf16) + x1n (fp32 native)
  gemm_bf16_kernel<true><<<dim3(GN / 128, GM / 256), 256, 0, stream>>>(xT, waT, x1t, x1n);

  // 4) x2 = x1t @ wa  -> x2n (fp32 native) only
  gemm_bf16_kernel<false><<<dim3(GN / 128, GM / 256), 256, 0, stream>>>(x1t, waT, (__bf16*)nullptr, x2n);

  // 5) channel-mix mlp + bias
  mlp_kernel<<<dim3((GB * GN * GL) / 256), 256, 0, stream>>>(x, x1n, x2n, mw, mb, out);
}
